// GroupedQueryAttention_68685116998065
// MI455X (gfx1250) — compile-verified, matrix operand order checked by reading, not run on hardware
//
#include <hip/hip_runtime.h>
#include <math.h>

// Problem constants (match reference)
#define B_   2
#define T_   2048
#define D_   2048
#define NH_  16
#define NKV_ 4
#define HD_  128
#define GRP_ (NH_/NKV_)
#define E_   (NH_*HD_)    // 2048
#define RD_  64

typedef unsigned short u16;
typedef __attribute__((ext_vector_type(16))) __bf16 v16bf;
typedef __attribute__((ext_vector_type(8)))  __bf16 v8bf;
typedef __attribute__((ext_vector_type(8)))  float  v8f;

union Frag { v16bf v; v8bf h[2]; };

__device__ __forceinline__ u16 f2bf(float f) {
  unsigned int u = __float_as_uint(f);
  unsigned int r = (u + 0x7FFFu + ((u >> 16) & 1u)) >> 16;  // RNE
  return (u16)r;
}
__device__ __forceinline__ v8bf ld8g(const u16* p) { return *(const v8bf*)p; }

// ---------------------------------------------------------------------------
// 1) RMSNorm: x (row of D f32) -> h bf16
// ---------------------------------------------------------------------------
__global__ void rmsnorm_kernel(const float* __restrict__ x,
                               const float* __restrict__ w,
                               u16* __restrict__ h) {
  __shared__ float red[8];
  const int row = blockIdx.x;
  const int tid = threadIdx.x;
  const float* xr = x + (size_t)row * D_;
  float ss = 0.f;
  for (int i = tid; i < D_; i += 256) { float v = xr[i]; ss += v * v; }
  #pragma unroll
  for (int m = 16; m >= 1; m >>= 1) ss += __shfl_xor(ss, m, 32);
  if ((tid & 31) == 0) red[tid >> 5] = ss;
  __syncthreads();
  float tot = 0.f;
  #pragma unroll
  for (int i = 0; i < 8; i++) tot += red[i];
  const float scale = rsqrtf(tot / (float)D_ + 1e-6f);
  for (int i = tid; i < D_; i += 256)
    h[(size_t)row * D_ + i] = f2bf(xr[i] * scale * w[i]);
}

// ---------------------------------------------------------------------------
// 2) f32 -> bf16 convert (weights)
// ---------------------------------------------------------------------------
__global__ void cvt_bf16_kernel(const float* __restrict__ src,
                                u16* __restrict__ dst, int n) {
  int i = blockIdx.x * 256 + threadIdx.x;
  if (i < n) dst[i] = f2bf(src[i]);
}

// ---------------------------------------------------------------------------
// 3) Generic GEMM: C[M,N] f32 = A[M,K] bf16 * W[N,K]^T bf16
//    One wave computes a 32x64 tile: 2 A-fragments x 4 B-fragments,
//    8 accumulators -> 12 b128 loads per 8 WMMAs (1.5 loads/WMMA).
//    Fragment layout per CDNA5 ISA 16-bit A/B tables:
//    lane = (m&15) + 16*(k-half), each lane holds two contiguous 8-elem runs.
// ---------------------------------------------------------------------------
__global__ __launch_bounds__(256) void gemm_bf16_wmma(
    const u16* __restrict__ A, const u16* __restrict__ W,
    float* __restrict__ C, int M, int N, int K) {
  const int wave = blockIdx.x * (blockDim.x >> 5) + (threadIdx.x >> 5);
  const int lane = threadIdx.x & 31;
  const int mtiles = M >> 5;                 // 32-row tiles
  const int mtile = wave % mtiles;
  const int ntile = wave / mtiles;
  if (ntile >= (N >> 6)) return;

  const int m    = lane & 15;
  const int koff = (lane >> 4) * 8;

  const u16* arow0 = A + (size_t)(mtile * 32 +  0 + m) * K;
  const u16* arow1 = A + (size_t)(mtile * 32 + 16 + m) * K;
  const u16* w0 = W + (size_t)(ntile * 64 +  0 + m) * K;
  const u16* w1 = W + (size_t)(ntile * 64 + 16 + m) * K;
  const u16* w2 = W + (size_t)(ntile * 64 + 32 + m) * K;
  const u16* w3 = W + (size_t)(ntile * 64 + 48 + m) * K;

  v8f acc[2][4];
  #pragma unroll
  for (int i = 0; i < 2; i++)
    #pragma unroll
    for (int j = 0; j < 4; j++) acc[i][j] = (v8f){};

  for (int k = 0; k < K; k += 32) {
    Frag a0, a1;
    a0.h[0] = ld8g(arow0 + k + koff); a0.h[1] = ld8g(arow0 + k + 16 + koff);
    a1.h[0] = ld8g(arow1 + k + koff); a1.h[1] = ld8g(arow1 + k + 16 + koff);
    Frag b0, b1, b2, b3;
    b0.h[0] = ld8g(w0 + k + koff); b0.h[1] = ld8g(w0 + k + 16 + koff);
    b1.h[0] = ld8g(w1 + k + koff); b1.h[1] = ld8g(w1 + k + 16 + koff);
    b2.h[0] = ld8g(w2 + k + koff); b2.h[1] = ld8g(w2 + k + 16 + koff);
    b3.h[0] = ld8g(w3 + k + koff); b3.h[1] = ld8g(w3 + k + 16 + koff);
    acc[0][0] = __builtin_amdgcn_wmma_f32_16x16x32_bf16(false, a0.v, false, b0.v, (short)0, acc[0][0], false, false);
    acc[0][1] = __builtin_amdgcn_wmma_f32_16x16x32_bf16(false, a0.v, false, b1.v, (short)0, acc[0][1], false, false);
    acc[0][2] = __builtin_amdgcn_wmma_f32_16x16x32_bf16(false, a0.v, false, b2.v, (short)0, acc[0][2], false, false);
    acc[0][3] = __builtin_amdgcn_wmma_f32_16x16x32_bf16(false, a0.v, false, b3.v, (short)0, acc[0][3], false, false);
    acc[1][0] = __builtin_amdgcn_wmma_f32_16x16x32_bf16(false, a1.v, false, b0.v, (short)0, acc[1][0], false, false);
    acc[1][1] = __builtin_amdgcn_wmma_f32_16x16x32_bf16(false, a1.v, false, b1.v, (short)0, acc[1][1], false, false);
    acc[1][2] = __builtin_amdgcn_wmma_f32_16x16x32_bf16(false, a1.v, false, b2.v, (short)0, acc[1][2], false, false);
    acc[1][3] = __builtin_amdgcn_wmma_f32_16x16x32_bf16(false, a1.v, false, b3.v, (short)0, acc[1][3], false, false);
  }

  const int hi = lane >> 4;
  #pragma unroll
  for (int i = 0; i < 2; i++) {
    #pragma unroll
    for (int r = 0; r < 8; r++) {
      const int mm = mtile * 32 + i * 16 + r + 8 * hi;
      const int nn = ntile * 64 + (lane & 15);
      float* crow = C + (size_t)mm * N + nn;
      crow[ 0] = acc[i][0][r];
      crow[16] = acc[i][1][r];
      crow[32] = acc[i][2][r];
      crow[48] = acc[i][3][r];
    }
  }
}

// ---------------------------------------------------------------------------
// 4) RoPE (first 64 dims) + layout pack: src (b,t,nh,HD) f32 -> dst (b,nh,t,HD) bf16
// ---------------------------------------------------------------------------
__global__ void rope_pack_kernel(const float* __restrict__ src,
                                 const float* __restrict__ cosT,
                                 const float* __restrict__ sinT,
                                 u16* __restrict__ dst, int nheads) {
  const int row = blockIdx.x;              // (b*T + t)*nheads + h
  const int d   = threadIdx.x;             // 0..127
  const int h   = row % nheads;
  const int t   = (row / nheads) % T_;
  const int b   = row / (nheads * T_);
  const float* s = src + (size_t)row * HD_;
  float v = s[d];
  if (d < RD_) {
    const float c  = cosT[t * RD_ + d];
    const float sn = sinT[t * RD_ + d];
    const float rot = (d < RD_ / 2) ? -s[d + RD_ / 2] : s[d - RD_ / 2];
    v = v * c + rot * sn;
  }
  dst[(((size_t)b * nheads + h) * T_ + t) * HD_ + d] = f2bf(v);
}

// 5) V pack transposed: src (b,t,nkv,HD) f32 -> vt (b,nkv,HD,T) bf16
__global__ void v_transpose_kernel(const float* __restrict__ src,
                                   u16* __restrict__ vt) {
  const int row = blockIdx.x;              // (b*T + t)*NKV + kv
  const int d   = threadIdx.x;
  const int kv  = row % NKV_;
  const int t   = (row / NKV_) % T_;
  const int b   = row / (NKV_ * T_);
  const float v = src[(size_t)row * HD_ + d];
  vt[(((size_t)b * NKV_ + kv) * HD_ + d) * T_ + t] = f2bf(v);
}

// ---------------------------------------------------------------------------
// 6) Flash attention + XSA. One wave per (b, h, 16-query tile).
//    Key tiles of 32.  S via 2x4 WMMAs; P->LDS->A-frag; O via 8 WMMAs.
// ---------------------------------------------------------------------------
__global__ __launch_bounds__(128) void attn_kernel(
    const u16* __restrict__ qb,   // (b, nh, t, HD) bf16
    const u16* __restrict__ kb,   // (b, nkv, t, HD) bf16
    const u16* __restrict__ vtb,  // (b, nkv, HD, T) bf16
    const float* __restrict__ vf, // (b, t, nkv, HD) f32
    u16* __restrict__ ob) {       // (b, t, nh*HD) bf16
  __shared__ u16   Plds[4][16 * 32];
  __shared__ float Vlds[4][16 * HD_];

  const int wib  = threadIdx.x >> 5;
  const int lane = threadIdx.x & 31;
  const int wave = blockIdx.x * 4 + wib;
  const int qt = wave % (T_ / 16);
  const int h  = (wave / (T_ / 16)) % NH_;
  const int b  = wave / ((T_ / 16) * NH_);
  const int q0 = qt * 16;
  const int kv = h / GRP_;

  const int m    = lane & 15;
  const int hi   = lane >> 4;
  const int koff = hi * 8;

  // Q fragments (4 slices over HD=128)
  const u16* qrow = qb + ((((size_t)b * NH_ + h) * T_) + q0 + m) * HD_;
  Frag qf[4];
  #pragma unroll
  for (int f = 0; f < 4; f++) {
    qf[f].h[0] = ld8g(qrow + f * 32 + koff);
    qf[f].h[1] = ld8g(qrow + f * 32 + 16 + koff);
  }

  v8f o[8];
  float rm[8], rl[8];
  #pragma unroll
  for (int f = 0; f < 8; f++) { o[f] = (v8f){}; }
  #pragma unroll
  for (int r = 0; r < 8; r++) { rm[r] = -3.0e38f; rl[r] = 0.f; }

  const float sm_scale = 0.08838834764831845f;  // 1/sqrt(128)
  const u16* kbase = kb + (((size_t)b * NKV_ + kv) * T_) * HD_;
  const u16* vbase = vtb + (((size_t)b * NKV_ + kv) * HD_) * (size_t)T_;
  u16* P = Plds[wib];

  for (int j0 = 0; j0 <= q0 + 15; j0 += 32) {
    // ---- S = Q * K^T, two 16-key halves ----
    v8f s0 = {}, s1 = {};
    {
      const u16* kr0 = kbase + (size_t)(j0 + m) * HD_;
      const u16* kr1 = kbase + (size_t)(j0 + 16 + m) * HD_;
      #pragma unroll
      for (int f = 0; f < 4; f++) {
        Frag kb0, kb1;
        kb0.h[0] = ld8g(kr0 + f * 32 + koff); kb0.h[1] = ld8g(kr0 + f * 32 + 16 + koff);
        kb1.h[0] = ld8g(kr1 + f * 32 + koff); kb1.h[1] = ld8g(kr1 + f * 32 + 16 + koff);
        s0 = __builtin_amdgcn_wmma_f32_16x16x32_bf16(false, qf[f].v, false, kb0.v, (short)0, s0, false, false);
        s1 = __builtin_amdgcn_wmma_f32_16x16x32_bf16(false, qf[f].v, false, kb1.v, (short)0, s1, false, false);
      }
    }
    // ---- scale + causal mask + online softmax ----
    #pragma unroll
    for (int r = 0; r < 8; r++) {
      const int qpos = q0 + r + 8 * hi;
      const int key0 = j0 + m;
      const int key1 = j0 + 16 + m;
      float a0 = s0[r] * sm_scale;
      float a1 = s1[r] * sm_scale;
      if (key0 > qpos) a0 = -1.0e30f;
      if (key1 > qpos) a1 = -1.0e30f;

      float mx = fmaxf(a0, a1);
      #pragma unroll
      for (int d2 = 1; d2 < 16; d2 <<= 1) mx = fmaxf(mx, __shfl_xor(mx, d2, 32));
      const float nm = fmaxf(rm[r], mx);
      const float sc = __expf(rm[r] - nm);
      const float p0 = __expf(a0 - nm);
      const float p1 = __expf(a1 - nm);
      float ps = p0 + p1;
      #pragma unroll
      for (int d2 = 1; d2 < 16; d2 <<= 1) ps += __shfl_xor(ps, d2, 32);
      rl[r] = rl[r] * sc + ps;
      rm[r] = nm;
      #pragma unroll
      for (int f = 0; f < 8; f++) o[f][r] *= sc;
      // P -> LDS (16x32 bf16, row-major)
      const int mm = r + 8 * hi;
      P[mm * 32 + m]      = f2bf(p0);
      P[mm * 32 + 16 + m] = f2bf(p1);
    }
    asm volatile("s_wait_dscnt 0" ::: "memory");
    // P as A-fragment
    Frag pf;
    pf.h[0] = *(const v8bf*)(P + m * 32 + koff);
    pf.h[1] = *(const v8bf*)(P + m * 32 + 16 + koff);
    // ---- O += P * V (8 dim-slices of 16) ----
    #pragma unroll
    for (int f = 0; f < 8; f++) {
      const u16* vr = vbase + (size_t)(f * 16 + m) * T_ + j0;
      Frag vfr;
      vfr.h[0] = ld8g(vr + koff);
      vfr.h[1] = ld8g(vr + 16 + koff);
      o[f] = __builtin_amdgcn_wmma_f32_16x16x32_bf16(false, pf.v, false, vfr.v, (short)0, o[f], false, false);
    }
  }

  // ---- normalize ----
  #pragma unroll
  for (int r = 0; r < 8; r++) {
    const float inv = 1.0f / rl[r];
    #pragma unroll
    for (int f = 0; f < 8; f++) o[f][r] *= inv;
  }

  // ---- stage V(f32) tile for XSA: rows q0..q0+15, 128 dims ----
  float* VL = Vlds[wib];
  for (int it = lane; it < 512; it += 32) {
    const int rr = it >> 5;          // row 0..15
    const int cc = (it & 31) * 4;    // col (float4)
    const float* src = vf + (((size_t)(b * T_ + q0 + rr) * NKV_ + kv) * HD_) + cc;
    *(float4*)(VL + rr * HD_ + cc) = *(const float4*)src;
  }
  asm volatile("s_wait_loadcnt 0\n\ts_wait_dscnt 0" ::: "memory");

  // ---- XSA: out -= (out.v) v / (v.v + 1e-6) with v at the query position ----
  #pragma unroll
  for (int r = 0; r < 8; r++) {
    const int mm = r + 8 * hi;
    float vv[8];
    float dot = 0.f, vns = 0.f;
    #pragma unroll
    for (int f = 0; f < 8; f++) {
      const float x = VL[mm * HD_ + f * 16 + m];
      vv[f] = x;
      dot += o[f][r] * x;
      vns += x * x;
    }
    #pragma unroll
    for (int d2 = 1; d2 < 16; d2 <<= 1) {
      dot += __shfl_xor(dot, d2, 32);
      vns += __shfl_xor(vns, d2, 32);
    }
    const float g = dot / (vns + 1e-6f);
    #pragma unroll
    for (int f = 0; f < 8; f++) o[f][r] -= g * vv[f];
  }

  // ---- write attn output (b, t, h*HD + d) bf16 ----
  #pragma unroll
  for (int r = 0; r < 8; r++) {
    const int mm = r + 8 * hi;
    u16* orow = ob + ((size_t)b * T_ + (q0 + mm)) * E_ + h * HD_;
    #pragma unroll
    for (int f = 0; f < 8; f++) orow[f * 16 + m] = f2bf(o[f][r]);
  }
}

// ---------------------------------------------------------------------------
// Launcher
// ---------------------------------------------------------------------------
extern "C" void kernel_launch(void* const* d_in, const int* in_sizes, int n_in,
                              void* d_out, int out_size, void* d_ws, size_t ws_size,
                              hipStream_t stream) {
  const float* x    = (const float*)d_in[0];
  const float* cosT = (const float*)d_in[1];
  const float* sinT = (const float*)d_in[2];
  const float* wn   = (const float*)d_in[3];
  const float* wq   = (const float*)d_in[4];
  const float* wk   = (const float*)d_in[5];
  const float* wv   = (const float*)d_in[6];
  const float* wo   = (const float*)d_in[7];
  float* out = (float*)d_out;

  // workspace carve (256B aligned)
  char* ws = (char*)d_ws;
  size_t off = 0;
  #define CARVE(ptr, type, count) \
    type* ptr = (type*)(ws + off); off = (off + (size_t)(count) * sizeof(type) + 255) & ~(size_t)255;

  CARVE(h_bf,   u16,   (size_t)B_ * T_ * D_);          // 16 MB
  CARVE(wq_bf,  u16,   (size_t)E_ * D_);               //  8 MB
  CARVE(wk_bf,  u16,   (size_t)NKV_ * HD_ * D_);       //  2 MB
  CARVE(wv_bf,  u16,   (size_t)NKV_ * HD_ * D_);       //  2 MB
  CARVE(wo_bf,  u16,   (size_t)D_ * E_);               //  8 MB
  CARVE(q_f,    float, (size_t)B_ * T_ * E_);          // 32 MB
  CARVE(k_f,    float, (size_t)B_ * T_ * NKV_ * HD_);  //  8 MB
  CARVE(v_f,    float, (size_t)B_ * T_ * NKV_ * HD_);  //  8 MB
  CARVE(q_bf,   u16,   (size_t)B_ * NH_ * T_ * HD_);   // 16 MB
  CARVE(k_bf,   u16,   (size_t)B_ * NKV_ * T_ * HD_);  //  4 MB
  CARVE(vt_bf,  u16,   (size_t)B_ * NKV_ * HD_ * T_);  //  4 MB
  CARVE(attn_bf,u16,   (size_t)B_ * T_ * E_);          // 16 MB
  (void)ws_size; (void)n_in; (void)in_sizes; (void)out_size;

  const int rows = B_ * T_;  // 4096

  // 1) RMSNorm -> bf16
  rmsnorm_kernel<<<rows, 256, 0, stream>>>(x, wn, h_bf);

  // 2) weight converts
  cvt_bf16_kernel<<<(E_ * D_ + 255) / 256, 256, 0, stream>>>(wq, wq_bf, E_ * D_);
  cvt_bf16_kernel<<<(NKV_ * HD_ * D_ + 255) / 256, 256, 0, stream>>>(wk, wk_bf, NKV_ * HD_ * D_);
  cvt_bf16_kernel<<<(NKV_ * HD_ * D_ + 255) / 256, 256, 0, stream>>>(wv, wv_bf, NKV_ * HD_ * D_);
  cvt_bf16_kernel<<<(D_ * E_ + 255) / 256, 256, 0, stream>>>(wo, wo_bf, D_ * E_);

  // 3) Q/K/V projections  (waves = (M/32)*(N/64); 8 waves per block)
  {
    int wavesQ = (rows / 32) * (E_ / 64);          // 128*32 = 4096
    gemm_bf16_wmma<<<wavesQ / 8, 256, 0, stream>>>(h_bf, wq_bf, q_f, rows, E_, D_);
    int wavesKV = (rows / 32) * ((NKV_ * HD_) / 64); // 128*8 = 1024
    gemm_bf16_wmma<<<wavesKV / 8, 256, 0, stream>>>(h_bf, wk_bf, k_f, rows, NKV_ * HD_, D_);
    gemm_bf16_wmma<<<wavesKV / 8, 256, 0, stream>>>(h_bf, wv_bf, v_f, rows, NKV_ * HD_, D_);
  }

  // 4) RoPE + pack
  rope_pack_kernel<<<rows * NH_, HD_, 0, stream>>>(q_f, cosT, sinT, q_bf, NH_);
  rope_pack_kernel<<<rows * NKV_, HD_, 0, stream>>>(k_f, cosT, sinT, k_bf, NKV_);
  v_transpose_kernel<<<rows * NKV_, HD_, 0, stream>>>(v_f, vt_bf);

  // 5) attention (+XSA): one wave per (b,h,qtile); 4 waves/block
  {
    int waves = B_ * NH_ * (T_ / 16);  // 4096
    attn_kernel<<<waves / 4, 128, 0, stream>>>(q_bf, k_bf, vt_bf, v_f, attn_bf);
  }

  // 6) output projection: out = attn * wo^T
  {
    int waves = (rows / 32) * (D_ / 64);           // 4096
    gemm_bf16_wmma<<<waves / 8, 256, 0, stream>>>(attn_bf, wo_bf, out, rows, D_, E_);
  }
}
